// MotionFeatureExtraction_84645215470250
// MI455X (gfx1250) — matrix operand order repaired and checked by reading, not compiled, not run on hardware
//
#include <hip/hip_runtime.h>
#include <hip/hip_bf16.h>
#include <stdint.h>

// ---------------------------------------------------------------------------
// Sparse 3x3x3 submanifold conv (2 layers) for MI455X / gfx1250.
// bf16 activations+weights, f32 accumulation via v_wmma_f32_16x16x32_bf16.
// Dense 130^3 key->index table replaces the reference sort/searchsorted.
// Layer weights staged in LDS (216KB / 108KB of the 320KB WGP LDS); one wave
// computes a full 16xCOUT output tile. All 27 neighbor indices prefetched,
// A-fragment gathers double-buffered across taps to overlap L2 latency with
// the 8 WMMAs per tap.
// ---------------------------------------------------------------------------

typedef __attribute__((ext_vector_type(16))) __bf16 v16bf;
typedef __attribute__((ext_vector_type(8)))  __bf16 v8bf;
typedef __attribute__((ext_vector_type(8)))  float  v8f;

#define GRIDB   130                   // 128 + 2*stride, stride == 1
#define TBL_N   (GRIDB*GRIDB*GRIDB)   // 2,197,000 entries
#define WPW     16                    // waves (M-tiles) per block

__device__ __forceinline__ v16bf cat16(v8bf lo, v8bf hi) {
  return __builtin_shufflevector(lo, hi, 0,1,2,3,4,5,6,7,8,9,10,11,12,13,14,15);
}

__device__ __forceinline__ int enc_key(const int* c) {
  // ((c0*B + c1 + s)*B + c2 + s)*B + c3 + s  with s=1, B=130
  return ((c[0]*GRIDB + c[1] + 1)*GRIDB + c[2] + 1)*GRIDB + c[3] + 1;
}

// A fragment for one tap: K=0..63 of one gathered row, per-lane layout.
struct AFrag { v8bf lo0, hi0, lo1, hi1; };

__device__ __forceinline__ AFrag loadA(const __bf16* __restrict__ X, int nb, int half) {
  AFrag a;
  a.lo0 = (v8bf){}; a.hi0 = (v8bf){}; a.lo1 = (v8bf){}; a.hi1 = (v8bf){};
  if (nb >= 0) {
    const v8bf* xa = (const v8bf*)(X + (size_t)nb * 64);
    a.lo0 = xa[half];        // K =  0 + half*8 + 0..7
    a.hi0 = xa[2 + half];    // K = 16 + half*8 + 0..7
    a.lo1 = xa[4 + half];    // K = 32 + half*8 + 0..7
    a.hi1 = xa[6 + half];    // K = 48 + half*8 + 0..7
  }
  return a;
}

// ---------------------------------------------------------------- setup ----
__global__ void k_init_table(int* __restrict__ tbl, int n) {
  int i = blockIdx.x * blockDim.x + threadIdx.x;
  if (i < n) tbl[i] = -1;
}

__global__ void k_build_keys(const int* __restrict__ coords,
                             int* __restrict__ tbl, int* __restrict__ keys, int n) {
  int i = blockIdx.x * blockDim.x + threadIdx.x;
  if (i >= n) return;
  int key = enc_key(coords + 4*i);
  keys[i] = key;
  tbl[key] = i;
}

__global__ void k_zero16(uint4* __restrict__ p, int n16) {
  int i = blockIdx.x * blockDim.x + threadIdx.x;
  if (i < n16) { uint4 z; z.x = z.y = z.z = z.w = 0u; p[i] = z; }
}

// one thread per (point, channel); channel = tid & 31
__global__ void k_scatter_feats(const int* __restrict__ coords,
                                const float* __restrict__ feats,
                                const int* __restrict__ tbl,
                                __bf16* __restrict__ X,
                                int* __restrict__ idx_out,
                                int n, int colOff) {
  int tid = blockIdx.x * blockDim.x + threadIdx.x;
  int i = tid >> 5, c = tid & 31;
  if (i >= n) return;
  int r = tbl[enc_key(coords + 4*i)];
  if (r < 0) return;
  X[(size_t)r*64 + colOff + c] = (__bf16)feats[(size_t)i*32 + c];
  if (idx_out && c == 0) idx_out[i] = r;
}

// Pack W (27,64,COUT) f32 -> per-lane B-fragment bf16 layout:
// PB[(((k*2+kc)*NT + nt)*32 + lane)*16 + t] = W[k][kc*32 + (t>>3)*16 + (lane>>4)*8 + (t&7)][nt*16 + (lane&15)]
__global__ void k_pack_w(const float* __restrict__ W, __bf16* __restrict__ PB,
                         int COUT, int NT, int total) {
  int idx = blockIdx.x * blockDim.x + threadIdx.x;
  if (idx >= total) return;
  int t    = idx & 15;
  int u    = idx >> 4;
  int lane = u & 31;   u >>= 5;
  int nt   = u % NT;   u /= NT;
  int kc   = u & 1;
  int k    = u >> 1;
  int K    = kc*32 + ((t >> 3) * 16) + ((lane >> 4) * 8) + (t & 7);
  int col  = nt*16 + (lane & 15);
  PB[idx] = (__bf16)W[((size_t)k*64 + K)*COUT + col];
}

// ----------------------------------------------------------------- conv ----
template <int COUT, bool RELU, typename OT, bool GATHER>
__global__ __launch_bounds__(32*WPW)
void k_conv_wmma(const __bf16* __restrict__ X,
                 const __bf16* __restrict__ PB,
                 const float* __restrict__ bias,
                 const int*   __restrict__ tbl,
                 const int*   __restrict__ keys,
                 const int*   __restrict__ rows,   // idx_pred when GATHER
                 OT*          __restrict__ out,
                 int M) {
  constexpr int NT  = COUT / 16;
  constexpr int WSZ = 27 * 2 * NT * 32 * 16;       // packed weight elems (bf16)

  extern __shared__ __bf16 sw[];
  // cooperative stage of the whole layer's packed weights into LDS
  for (int i = threadIdx.x; i < WSZ / 8; i += blockDim.x)
    ((uint4*)sw)[i] = ((const uint4*)PB)[i];
  __syncthreads();

  const int lane = threadIdx.x & 31;
  const int wave = threadIdx.x >> 5;
  const int r    = lane & 15;                 // A row / D column within tile
  const int half = lane >> 4;                 // K-half (A/B), M-half (C/D)
  const int m16  = (blockIdx.x * WPW + wave) * 16;
  if (m16 >= M) return;                       // no further barriers

  int rl = m16 + r; if (rl >= M) rl = M - 1;
  const int row   = GATHER ? rows[rl] : rl;   // union-space row for this lane
  const int mykey = keys[row];

  // prefetch all 27 neighbor indices (independent loads, batched)
  int nbr[28];
#pragma unroll
  for (int k = 0; k < 27; ++k) {
    const int dx = k/9 - 1, dy = (k/3)%3 - 1, dz = k%3 - 1;
    nbr[k] = tbl[mykey + (dx*GRIDB + dy)*GRIDB + dz];
  }
  nbr[27] = -1;                               // pipeline sentinel -> zero frag

  v8f acc[NT];
#pragma unroll
  for (int n = 0; n < NT; ++n) acc[n] = (v8f){};

  AFrag cur = loadA(X, nbr[0], half);
#pragma unroll
  for (int k = 0; k < 27; ++k) {
    AFrag nxt = loadA(X, nbr[k + 1], half);   // overlap gather with WMMAs
    const v16bf A0 = cat16(cur.lo0, cur.hi0); // K chunk 0..31
    const v16bf A1 = cat16(cur.lo1, cur.hi1); // K chunk 32..63
#pragma unroll
    for (int nt = 0; nt < NT; ++nt) {
      const v8bf* pb = (const v8bf*)(sw + (((k*2 + 0)*NT + nt)*32 + lane) * 16);
      v16bf B = cat16(pb[0], pb[1]);
      acc[nt] = __builtin_amdgcn_wmma_f32_16x16x32_bf16(
          false, A0, false, B, (short)0, acc[nt], false, false);
    }
#pragma unroll
    for (int nt = 0; nt < NT; ++nt) {
      const v8bf* pb = (const v8bf*)(sw + (((k*2 + 1)*NT + nt)*32 + lane) * 16);
      v16bf B = cat16(pb[0], pb[1]);
      acc[nt] = __builtin_amdgcn_wmma_f32_16x16x32_bf16(
          false, A1, false, B, (short)0, acc[nt], false, false);
    }
    cur = nxt;
  }

#pragma unroll
  for (int nt = 0; nt < NT; ++nt) {
    const int col = nt*16 + r;
    const float bv = bias[col];
#pragma unroll
    for (int j = 0; j < 8; ++j) {
      const int orow = m16 + half*8 + j;      // C/D: VGPR j -> row half*8+j
      if (orow < M) {
        float v = acc[nt][j] + bv;
        if (RELU) v = fmaxf(v, 0.0f);
        out[(size_t)orow*COUT + col] = (OT)v;
      }
    }
  }
}

// ---------------------------------------------------------------- launch ---
extern "C" void kernel_launch(void* const* d_in, const int* in_sizes, int n_in,
                              void* d_out, int out_size, void* d_ws, size_t ws_size,
                              hipStream_t stream) {
  const int*   coords_ref   = (const int*)  d_in[0];
  const int*   coords_pred  = (const int*)  d_in[1];
  const int*   coords_union = (const int*)  d_in[2];
  const float* feats_ref    = (const float*)d_in[3];
  const float* feats_pred   = (const float*)d_in[4];
  const float* W0           = (const float*)d_in[5];
  const float* b0           = (const float*)d_in[6];
  const float* W1           = (const float*)d_in[7];
  const float* b1           = (const float*)d_in[8];

  const int nR = in_sizes[0] / 4;
  const int nP = in_sizes[1] / 4;
  const int nU = in_sizes[2] / 4;

  // carve workspace (all offsets 256B aligned)
  char* ws = (char*)d_ws;
  size_t off = 0;
  auto carve = [&](size_t bytes) {
    char* p = ws + off;
    off += (bytes + 255) & ~size_t(255);
    return p;
  };
  int*    tbl  = (int*)   carve((size_t)TBL_N * 4);
  int*    keys = (int*)   carve((size_t)nU * 4);
  int*    idxp = (int*)   carve((size_t)nP * 4);
  __bf16* X    = (__bf16*)carve((size_t)nU * 64 * 2);
  __bf16* Y    = (__bf16*)carve((size_t)nU * 64 * 2);
  __bf16* PB0  = (__bf16*)carve((size_t)27*2*4*32*16 * 2);
  __bf16* PB1  = (__bf16*)carve((size_t)27*2*2*32*16 * 2);
  (void)ws_size; (void)n_in; (void)out_size;

  const int B = 256;

  k_init_table<<<(TBL_N + B - 1)/B, B, 0, stream>>>(tbl, TBL_N);
  k_build_keys<<<(nU + B - 1)/B, B, 0, stream>>>(coords_union, tbl, keys, nU);

  const int nx16 = nU * 8;  // (nU*64 bf16)/8 per uint4
  k_zero16<<<(nx16 + B - 1)/B, B, 0, stream>>>((uint4*)X, nx16);

  k_scatter_feats<<<((size_t)nR*32 + B - 1)/B, B, 0, stream>>>(
      coords_ref, feats_ref, tbl, X, nullptr, nR, 0);
  k_scatter_feats<<<((size_t)nP*32 + B - 1)/B, B, 0, stream>>>(
      coords_pred, feats_pred, tbl, X, idxp, nP, 32);

  const int cnt0 = 27*2*4*32*16;   // 110592 bf16 = 216 KiB
  const int cnt1 = 27*2*2*32*16;   // 55296  bf16 = 108 KiB
  k_pack_w<<<(cnt0 + B - 1)/B, B, 0, stream>>>(W0, PB0, 64, 4, cnt0);
  k_pack_w<<<(cnt1 + B - 1)/B, B, 0, stream>>>(W1, PB1, 32, 2, cnt1);

  // layer 0: all union rows, Cout=64, relu, bf16 out. 16 M-tiles per block.
  {
    const int mtiles = (nU + 15) / 16;
    const int grid   = (mtiles + WPW - 1) / WPW;
    k_conv_wmma<64, true, __bf16, false><<<grid, 32*WPW, cnt0*2, stream>>>(
        X, PB0, b0, tbl, keys, nullptr, Y, nU);
  }

  // layer 1: only pred rows (tiled over idx_pred), Cout=32, f32 out.
  {
    const int mtiles = (nP + 15) / 16;
    const int grid   = (mtiles + WPW - 1) / WPW;
    k_conv_wmma<32, false, float, true><<<grid, 32*WPW, cnt1*2, stream>>>(
        Y, PB1, b1, tbl, keys, idxp, (float*)d_out, nP);
  }
}
